// CrystalGraphConvNet_17712445129487
// MI455X (gfx1250) — matrix operand order repaired and checked by reading, not compiled, not run on hardware
//
#include <hip/hip_runtime.h>
#include <stdint.h>

#define TRY_TDM 1

// ---------------- problem constants ----------------
#define CN      100000
#define CNPAD   100032          // N rounded up to 64 for the embed GEMM
#define CM      12
#define CORIG   92
#define CKE     96              // padded K for embedding (92 -> 96 = 3*32)
#define CNBR    41
#define CAF     64
#define C2AF    128
#define CKC     192             // padded K for conv (64+64+41=169 -> 192 = 6*32)
#define CHF     128
#define CN0     2000
#define CNCONV  3
#define GROUPS  (CN / 4)        // 25000 groups of 4 atoms (48 gemm rows each)

typedef __attribute__((ext_vector_type(16))) __bf16 v16bf;
typedef __attribute__((ext_vector_type(8)))  float  v8f;
typedef __attribute__((ext_vector_type(8)))  unsigned int v8u;
typedef __attribute__((ext_vector_type(4)))  unsigned int v4u;

union Frag16 { v16bf bf; v8u u8; v4u u4[2]; };

__device__ __forceinline__ unsigned short f2bf(float f) {
    unsigned int u = __float_as_uint(f);
    unsigned int r = u + 0x7fffu + ((u >> 16) & 1u);   // round-to-nearest-even
    return (unsigned short)(r >> 16);
}
__device__ __forceinline__ float softplus_f(float x) {
    return x > 20.f ? x : log1pf(__expf(x));
}
__device__ __forceinline__ float sigmoid_f(float x) {
    return 1.f / (1.f + __expf(-x));
}

// ---------------- prep kernels (fp32 -> padded bf16, weights transposed) ----
__global__ void k_prep_atomfea(const float* __restrict__ af, unsigned short* __restrict__ Abf) {
    for (long long e = (long long)blockIdx.x * blockDim.x + threadIdx.x;
         e < (long long)CNPAD * CKE; e += (long long)gridDim.x * blockDim.x) {
        int row = (int)(e / CKE), k = (int)(e % CKE);
        float v = (row < CN && k < CORIG) ? af[(long long)row * CORIG + k] : 0.f;
        Abf[e] = f2bf(v);
    }
}
__global__ void k_prep_embw(const float* __restrict__ w, unsigned short* __restrict__ WT) {
    for (int e = blockIdx.x * blockDim.x + threadIdx.x; e < CAF * CKE;
         e += gridDim.x * blockDim.x) {
        int c = e / CKE, k = e % CKE;
        WT[e] = f2bf(k < CORIG ? w[k * CAF + c] : 0.f);   // WT[col][K]
    }
}
__global__ void k_prep_fcw(const float* __restrict__ w, unsigned short* __restrict__ WT) {
    const int KIN = 2 * CAF + CNBR;                       // 169
    for (int e = blockIdx.x * blockDim.x + threadIdx.x; e < CNCONV * C2AF * CKC;
         e += gridDim.x * blockDim.x) {
        int l = e / (C2AF * CKC), rem = e % (C2AF * CKC);
        int c = rem / CKC, k = rem % CKC;
        WT[e] = f2bf(k < KIN ? w[(l * KIN + k) * C2AF + c] : 0.f); // [l][col][K]
    }
}

// ---------------- embedding GEMM: (NPAD x 96) @ (96 x 64), bf16 WMMA --------
__global__ __launch_bounds__(256) void k_embed(
    const unsigned short* __restrict__ Abf,   // [NPAD][96]
    const unsigned short* __restrict__ WT,    // [64][96]
    const float* __restrict__ bias,           // [64]
    float* __restrict__ atom_f32,             // [NPAD][64]
    unsigned short* __restrict__ atom_bf)     // [NPAD][64]
{
    __shared__ unsigned short As[64 * CKE];   // 12 KB A tile
    const int lane  = threadIdx.x & 31;
    const int wave  = threadIdx.x >> 5;
    const int ct    = wave & 3;               // col tile 0..3
    const int rhalf = wave >> 2;              // rows 0-31 / 32-63
    const int row0  = blockIdx.x * 64;

    // --- stage A tile into LDS: Tensor Data Mover if available -------------
#if TRY_TDM && __has_builtin(__builtin_amdgcn_tensor_load_to_lds) && __has_builtin(__builtin_amdgcn_s_wait_tensorcnt)
    if (threadIdx.x < 32) {
        typedef unsigned int u32x4 __attribute__((ext_vector_type(4)));
        typedef int          i32x8 __attribute__((ext_vector_type(8)));
        typedef int          i32x4 __attribute__((ext_vector_type(4)));
        unsigned long long ga = (unsigned long long)(uintptr_t)(Abf + (size_t)row0 * CKE);
        unsigned int lds_off  = (unsigned int)(uintptr_t)(&As[0]);
        // D# group0: count=1 | lds_addr | global_addr | type=2
        u32x4 g0;
        g0[0] = 1u;
        g0[1] = lds_off;
        g0[2] = (unsigned int)ga;
        g0[3] = ((unsigned int)(ga >> 32) & 0x1FFFFFFu) | 0x80000000u;
        // D# group1: data_size=2B; tensor_dim0=96; tensor_dim1=64;
        //            tile_dim0=96; tile_dim1=64; dim0_stride=96
        i32x8 g1;
        g1[0] = 0x10000;              // data_size = 1 (2 bytes)
        g1[1] = (int)(96u << 16);     // tensor_dim0[15:0] at bits[63:48]
        g1[2] = (int)(64u << 16);     // tensor_dim1[15:0] at bits[95:80]
        g1[3] = (int)(96u << 16);     // tile_dim0 at bits[127:112]
        g1[4] = 64;                   // tile_dim1
        g1[5] = 96;                   // tensor_dim0_stride lo32
        g1[6] = 0;
        g1[7] = 0;
        i32x4 z4 = {0, 0, 0, 0};
        i32x8 z8 = {0, 0, 0, 0, 0, 0, 0, 0};
        __builtin_amdgcn_tensor_load_to_lds(g0, g1, z4, z4, z8, 0);
        __builtin_amdgcn_s_wait_tensorcnt(0);
    }
    __syncthreads();
#else
    {
        const unsigned int* src = (const unsigned int*)(Abf + (size_t)row0 * CKE);
        unsigned int* dst = (unsigned int*)As;
        for (int e = threadIdx.x; e < 64 * CKE / 2; e += 256) dst[e] = src[e];
        __syncthreads();
    }
#endif

    const int col = ct * 16 + (lane & 15);
    v8f acc[2] = {};
    for (int kk = 0; kk < CKE; kk += 32) {
        const int kbB = kk + ((lane & 16) ? 16 : 0);
        Frag16 b; b.u8 = *(const v8u*)(WT + col * CKE + kbB);
        const int kbA = kk + ((lane & 16) ? 16 : 0);
#pragma unroll
        for (int t = 0; t < 2; ++t) {
            const int ar = rhalf * 32 + t * 16 + (lane & 15);
            Frag16 a;
            a.u4[0] = *(const v4u*)(As + ar * CKE + kbA);
            a.u4[1] = *(const v4u*)(As + ar * CKE + kbA + 8);
            acc[t] = __builtin_amdgcn_wmma_f32_16x16x32_bf16(
                false, a.bf, false, b.bf, (short)0, acc[t], false, false);
        }
    }
    const float bv = bias[col];
    if (row0 + 64 <= CN) {                    // fast path: no tail guards
#pragma unroll
        for (int t = 0; t < 2; ++t)
#pragma unroll
            for (int r = 0; r < 8; ++r) {
                int row = row0 + rhalf * 32 + t * 16 + r + ((lane & 16) ? 8 : 0);
                float v = acc[t][r] + bv;
                atom_f32[(size_t)row * CAF + col] = v;
                atom_bf[(size_t)row * CAF + col]  = f2bf(v);
            }
    } else {
#pragma unroll
        for (int t = 0; t < 2; ++t)
#pragma unroll
            for (int r = 0; r < 8; ++r) {
                int row = row0 + rhalf * 32 + t * 16 + r + ((lane & 16) ? 8 : 0);
                if (row < CN) {
                    float v = acc[t][r] + bv;
                    atom_f32[(size_t)row * CAF + col] = v;
                    atom_bf[(size_t)row * CAF + col]  = f2bf(v);
                }
            }
    }
}

// ---------------- conv layer helpers ---------------------------------------
// Stage concatenated A tile (48 rows = 4 atoms x 12 nbrs, K=192 bf16) into LDS
__device__ __forceinline__ void stage_conv_A(
    unsigned short* As, int* nbrs, int atom0,
    const unsigned short* __restrict__ atom_bf,
    const float* __restrict__ nbr_fea,
    const int* __restrict__ nbr_idx)
{
    if (threadIdx.x < 48) nbrs[threadIdx.x] = nbr_idx[atom0 * CM + threadIdx.x];
    __syncthreads();
    unsigned int* Aw = (unsigned int*)As;
    for (int e = threadIdx.x; e < 48 * (CKC / 2); e += 256) {
        int r = e / (CKC / 2);     // tile row 0..47
        int w = e % (CKC / 2);     // uint word 0..95  (2 bf16 each)
        int a = atom0 + r / CM;
        unsigned int val;
        if (w < 32) {                                        // self features
            val = *(const unsigned int*)(atom_bf + (size_t)a * CAF + w * 2);
        } else if (w < 64) {                                 // gathered neighbor
            val = *(const unsigned int*)(atom_bf + (size_t)nbrs[r] * CAF + (w - 32) * 2);
        } else {                                             // edge features (f32->bf16)
            long long row = (long long)atom0 * CM + r;
            int k0 = (w - 64) * 2;
            float f0 = (k0 < CNBR)     ? nbr_fea[row * CNBR + k0]     : 0.f;
            float f1 = (k0 + 1 < CNBR) ? nbr_fea[row * CNBR + k0 + 1] : 0.f;
            val = (unsigned int)f2bf(f0) | ((unsigned int)f2bf(f1) << 16);
        }
        Aw[e] = val;
    }
    __syncthreads();
}

// Load this wave's 6 loop-invariant B fragments (its 16-col tile of W^T)
__device__ __forceinline__ void load_B_frags(
    const unsigned short* __restrict__ WT, int lane, int col, Frag16 bfr[6])
{
#pragma unroll
    for (int s = 0; s < 6; ++s) {
        const int kbB = s * 32 + ((lane & 16) ? 16 : 0);
        bfr[s].u8 = *(const v8u*)(WT + col * CKC + kbB);
    }
}

// 48x192 (LDS bf16) @ resident B frags -> 3 v8f accumulators
__device__ __forceinline__ void conv_gemm(
    const unsigned short* As, const Frag16 bfr[6], int lane, v8f acc[3])
{
#pragma unroll
    for (int s = 0; s < 6; ++s) {
        const int kbA = s * 32 + ((lane & 16) ? 16 : 0);
#pragma unroll
        for (int t = 0; t < 3; ++t) {
            const int ar = t * 16 + (lane & 15);
            Frag16 a;
            a.u4[0] = *(const v4u*)(As + ar * CKC + kbA);
            a.u4[1] = *(const v4u*)(As + ar * CKC + kbA + 8);
            acc[t] = __builtin_amdgcn_wmma_f32_16x16x32_bf16(
                false, a.bf, false, bfr[s].bf, (short)0, acc[t], false, false);
        }
    }
}

// Pass 1: GEMM, accumulate per-channel sum / sumsq for BN1 (nothing stored)
__global__ __launch_bounds__(256) void k_conv_pass1(
    const unsigned short* __restrict__ atom_bf, const float* __restrict__ nbr_fea,
    const int* __restrict__ nbr_idx, const unsigned short* __restrict__ WT,
    const float* __restrict__ fb, float* __restrict__ sum1, float* __restrict__ sq1)
{
    __shared__ unsigned short As[48 * CKC];   // 18 KB
    __shared__ int nbrs[48];
    const int lane = threadIdx.x & 31;
    const int wave = threadIdx.x >> 5;
    const int col  = wave * 16 + (lane & 15);
    const float bv = fb[col];
    Frag16 bfr[6];
    load_B_frags(WT, lane, col, bfr);
    float psum = 0.f, psq = 0.f;
    for (int g = blockIdx.x; g < GROUPS; g += gridDim.x) {
        __syncthreads();
        stage_conv_A(As, nbrs, g * 4, atom_bf, nbr_fea, nbr_idx);
        v8f acc[3] = {};
        conv_gemm(As, bfr, lane, acc);
#pragma unroll
        for (int t = 0; t < 3; ++t)
#pragma unroll
            for (int r = 0; r < 8; ++r) {
                float v = acc[t][r] + bv;
                psum += v; psq += v * v;
            }
    }
    psum += __shfl_xor(psum, 16, 32);
    psq  += __shfl_xor(psq, 16, 32);
    if (lane < 16) { atomicAdd(&sum1[col], psum); atomicAdd(&sq1[col], psq); }
}

// Pass 2: GEMM again, BN1, sigmoid*softplus, sum over 12 neighbors,
//         accumulate BN2 stats, store summed[N][64]
__global__ __launch_bounds__(256) void k_conv_pass2(
    const unsigned short* __restrict__ atom_bf, const float* __restrict__ nbr_fea,
    const int* __restrict__ nbr_idx, const unsigned short* __restrict__ WT,
    const float* __restrict__ fb, const float* __restrict__ scale1,
    const float* __restrict__ shift1, float* __restrict__ summed,
    float* __restrict__ sum2, float* __restrict__ sq2)
{
    __shared__ unsigned short As[48 * CKC];   // 18 KB
    __shared__ int nbrs[48];
    __shared__ float gs[48 * C2AF];           // 24 KB normalized gated tile
    const int lane = threadIdx.x & 31;
    const int wave = threadIdx.x >> 5;
    const int col  = wave * 16 + (lane & 15);
    const float bv = fb[col];
    const float sc = scale1[col], sh = shift1[col];
    const int ac = threadIdx.x >> 6;          // atom-in-group 0..3
    const int cc = threadIdx.x & 63;          // output channel 0..63
    Frag16 bfr[6];
    load_B_frags(WT, lane, col, bfr);
    float psum = 0.f, psq = 0.f;
    for (int g = blockIdx.x; g < GROUPS; g += gridDim.x) {
        __syncthreads();
        stage_conv_A(As, nbrs, g * 4, atom_bf, nbr_fea, nbr_idx);
        v8f acc[3] = {};
        conv_gemm(As, bfr, lane, acc);
#pragma unroll
        for (int t = 0; t < 3; ++t)
#pragma unroll
            for (int r = 0; r < 8; ++r) {
                int row = t * 16 + r + ((lane & 16) ? 8 : 0);
                gs[row * C2AF + col] = (acc[t][r] + bv) * sc + sh;
            }
        __syncthreads();
        float s = 0.f;
#pragma unroll
        for (int m = 0; m < CM; ++m) {
            float f  = gs[(ac * CM + m) * C2AF + cc];
            float co = gs[(ac * CM + m) * C2AF + CAF + cc];
            s += sigmoid_f(f) * softplus_f(co);
        }
        summed[(size_t)(g * 4 + ac) * CAF + cc] = s;
        psum += s; psq += s * s;
    }
    __syncthreads();
    gs[threadIdx.x]       = psum;
    gs[256 + threadIdx.x] = psq;
    __syncthreads();
    if (threadIdx.x < 64) {
        float a = gs[threadIdx.x] + gs[threadIdx.x + 64] + gs[threadIdx.x + 128] + gs[threadIdx.x + 192];
        float b = gs[256 + threadIdx.x] + gs[256 + threadIdx.x + 64] +
                  gs[256 + threadIdx.x + 128] + gs[256 + threadIdx.x + 192];
        atomicAdd(&sum2[threadIdx.x], a);
        atomicAdd(&sq2[threadIdx.x], b);
    }
}

// ---------------- small kernels --------------------------------------------
__global__ void k_zero_stats(float* s1, float* q1, float* s2, float* q2) {
    int t = threadIdx.x;
    if (t < C2AF) { s1[t] = 0.f; q1[t] = 0.f; }
    if (t < CAF)  { s2[t] = 0.f; q2[t] = 0.f; }
}
__global__ void k_bn_finalize(const float* __restrict__ sum, const float* __restrict__ sq,
                              const float* __restrict__ g, const float* __restrict__ b,
                              float cnt, int nch, float* __restrict__ scale,
                              float* __restrict__ shift) {
    int i = threadIdx.x + blockIdx.x * blockDim.x;
    if (i < nch) {
        float mean = sum[i] / cnt;
        float var  = sq[i] / cnt - mean * mean;
        float s    = g[i] * rsqrtf(var + 1e-5f);
        scale[i] = s;
        shift[i] = b[i] - mean * s;
    }
}
__global__ void k_atom_update(const float* __restrict__ atom_in, const float* __restrict__ summed,
                              const float* __restrict__ scale2, const float* __restrict__ shift2,
                              float* __restrict__ atom_out, unsigned short* __restrict__ atom_out_bf) {
    for (int i = blockIdx.x * blockDim.x + threadIdx.x; i < CN * CAF;
         i += gridDim.x * blockDim.x) {
        int c = i & 63;
        float v = softplus_f(atom_in[i] + summed[i] * scale2[c] + shift2[c]);
        atom_out[i]    = v;
        atom_out_bf[i] = f2bf(v);
    }
}
__global__ void k_zero_pool(float* crys, float* cnt) {
    for (int i = blockIdx.x * blockDim.x + threadIdx.x; i < CN0 * CAF;
         i += gridDim.x * blockDim.x) crys[i] = 0.f;
    for (int i = blockIdx.x * blockDim.x + threadIdx.x; i < CN0;
         i += gridDim.x * blockDim.x) cnt[i] = 0.f;
}
__global__ void k_pool(const float* __restrict__ atom, const int* __restrict__ cidx,
                       float* __restrict__ crys, float* __restrict__ cnt) {
    for (int i = blockIdx.x * blockDim.x + threadIdx.x; i < CN * CAF;
         i += gridDim.x * blockDim.x)
        atomicAdd(&crys[cidx[i >> 6] * CAF + (i & 63)], atom[i]);
    for (int i = blockIdx.x * blockDim.x + threadIdx.x; i < CN;
         i += gridDim.x * blockDim.x)
        atomicAdd(&cnt[cidx[i]], 1.f);
}
// Heads: crys mean -> softplus -> @W(64x128)+b -> softplus -> @w(128x1)+b
__global__ __launch_bounds__(128) void k_heads(
    const float* __restrict__ crys, const float* __restrict__ cnt,
    const float* __restrict__ w1, const float* __restrict__ b1,
    const float* __restrict__ o1w, const float* __restrict__ o1b,
    const float* __restrict__ w2, const float* __restrict__ b2,
    const float* __restrict__ o2w, const float* __restrict__ o2b,
    float* __restrict__ out, int n0)
{
    __shared__ float sp[CAF];
    __shared__ float red[CHF];
    int c = blockIdx.x;
    if (c >= n0) return;
    int t = threadIdx.x;
    float inv = 1.f / fmaxf(cnt[c], 1.f);
    if (t < CAF) sp[t] = softplus_f(crys[c * CAF + t] * inv);
    __syncthreads();
    float d1 = b1[t], d2 = b2[t];
    for (int k = 0; k < CAF; ++k) {
        float s = sp[k];
        d1 += s * w1[k * CHF + t];
        d2 += s * w2[k * CHF + t];
    }
    float p1 = softplus_f(d1) * o1w[t];
    float p2 = softplus_f(d2) * o2w[t];
    red[t] = p1; __syncthreads();
    for (int s = 64; s > 0; s >>= 1) { if (t < s) red[t] += red[t + s]; __syncthreads(); }
    float r1 = red[0];
    __syncthreads();
    red[t] = p2; __syncthreads();
    for (int s = 64; s > 0; s >>= 1) { if (t < s) red[t] += red[t + s]; __syncthreads(); }
    if (t == 0) { out[c] = r1 + o1b[0]; out[n0 + c] = red[0] + o2b[0]; }
}

// ---------------- host launch ----------------------------------------------
extern "C" void kernel_launch(void* const* d_in, const int* in_sizes, int n_in,
                              void* d_out, int out_size, void* d_ws, size_t ws_size,
                              hipStream_t stream) {
    (void)in_sizes; (void)n_in; (void)out_size; (void)ws_size;
    const float* atom_fea = (const float*)d_in[1];
    const float* nbr_fea  = (const float*)d_in[2];
    const int*   nbr_idx  = (const int*)d_in[3];
    const int*   cidx     = (const int*)d_in[4];
    const float* emb_w    = (const float*)d_in[5];
    const float* emb_b    = (const float*)d_in[6];
    const float* fc_w     = (const float*)d_in[7];
    const float* fc_b     = (const float*)d_in[8];
    const float* bn1_g    = (const float*)d_in[9];
    const float* bn1_b    = (const float*)d_in[10];
    const float* bn2_g    = (const float*)d_in[11];
    const float* bn2_b    = (const float*)d_in[12];
    const float* c2f1_w   = (const float*)d_in[13];
    const float* c2f1_b   = (const float*)d_in[14];
    const float* out1_w   = (const float*)d_in[15];
    const float* out1_b   = (const float*)d_in[16];
    const float* c2f2_w   = (const float*)d_in[17];
    const float* c2f2_b   = (const float*)d_in[18];
    const float* out2_w   = (const float*)d_in[19];
    const float* out2_b   = (const float*)d_in[20];
    float* out = (float*)d_out;

    char* ws = (char*)d_ws;
    size_t off = 0;
    auto take = [&](size_t bytes) -> char* {
        off = (off + 255) & ~(size_t)255;
        char* p = ws + off;
        off += bytes;
        return p;
    };
    unsigned short* Abf     = (unsigned short*)take((size_t)CNPAD * CKE * 2);
    unsigned short* embT    = (unsigned short*)take((size_t)CAF * CKE * 2);
    unsigned short* fcT     = (unsigned short*)take((size_t)CNCONV * C2AF * CKC * 2);
    float* atomA            = (float*)take((size_t)CNPAD * CAF * 4);
    float* atomB            = (float*)take((size_t)CNPAD * CAF * 4);
    unsigned short* atomAbf = (unsigned short*)take((size_t)CNPAD * CAF * 2);
    unsigned short* atomBbf = (unsigned short*)take((size_t)CNPAD * CAF * 2);
    float* summed           = (float*)take((size_t)CN * CAF * 4);
    float* sum1   = (float*)take(C2AF * 4);
    float* sq1    = (float*)take(C2AF * 4);
    float* scale1 = (float*)take(C2AF * 4);
    float* shift1 = (float*)take(C2AF * 4);
    float* sum2   = (float*)take(CAF * 4);
    float* sq2    = (float*)take(CAF * 4);
    float* scale2 = (float*)take(CAF * 4);
    float* shift2 = (float*)take(CAF * 4);
    float* crys   = (float*)take((size_t)CN0 * CAF * 4);
    float* cnt    = (float*)take((size_t)CN0 * 4);

    k_prep_atomfea<<<2048, 256, 0, stream>>>(atom_fea, Abf);
    k_prep_embw<<<24, 256, 0, stream>>>(emb_w, embT);
    k_prep_fcw<<<288, 256, 0, stream>>>(fc_w, fcT);
    k_embed<<<CNPAD / 64, 256, 0, stream>>>(Abf, embT, emb_b, atomA, atomAbf);

    float* a_in = atomA;  unsigned short* abf_in = atomAbf;
    float* a_out = atomB; unsigned short* abf_out = atomBbf;
    for (int l = 0; l < CNCONV; ++l) {
        const unsigned short* WT = fcT + (size_t)l * C2AF * CKC;
        k_zero_stats<<<1, 128, 0, stream>>>(sum1, sq1, sum2, sq2);
        k_conv_pass1<<<2048, 256, 0, stream>>>(abf_in, nbr_fea, nbr_idx, WT,
                                               fc_b + l * C2AF, sum1, sq1);
        k_bn_finalize<<<1, 128, 0, stream>>>(sum1, sq1, bn1_g + l * C2AF, bn1_b + l * C2AF,
                                             (float)CN * (float)CM, C2AF, scale1, shift1);
        k_conv_pass2<<<2048, 256, 0, stream>>>(abf_in, nbr_fea, nbr_idx, WT,
                                               fc_b + l * C2AF, scale1, shift1,
                                               summed, sum2, sq2);
        k_bn_finalize<<<1, 128, 0, stream>>>(sum2, sq2, bn2_g + l * CAF, bn2_b + l * CAF,
                                             (float)CN, CAF, scale2, shift2);
        k_atom_update<<<2048, 256, 0, stream>>>(a_in, summed, scale2, shift2, a_out, abf_out);
        float* tf = a_in; a_in = a_out; a_out = tf;
        unsigned short* tb = abf_in; abf_in = abf_out; abf_out = tb;
    }
    k_zero_pool<<<512, 256, 0, stream>>>(crys, cnt);
    k_pool<<<2048, 256, 0, stream>>>(a_in, cidx, crys, cnt);
    k_heads<<<CN0, 128, 0, stream>>>(crys, cnt, c2f1_w, c2f1_b, out1_w, out1_b,
                                     c2f2_w, c2f2_b, out2_w, out2_b, out, CN0);
}
